// SmartPostProcess_25580825215219
// MI455X (gfx1250) — matrix operand order repaired and checked by reading, not compiled, not run on hardware
//
#include <hip/hip_runtime.h>
#include <hip/hip_bf16.h>

// ---------------------------------------------------------------------------
// SmartPostProcess for MI455X (gfx1250, wave32).
// Per-sigma {WMMA separable Gaussian blur, Hessian+eig+surfaceness,
// max-normalize}, max over sigmas, hysteresis (LDS-tile fixpoint with TDM
// tile DMA on interior tiles), z-only closing.
// Whole 16.4MB volume is L2-resident on MI455X (192MB L2).
// ---------------------------------------------------------------------------

#define NVOX   160
#define SLICE  (160 * 160)          // 25600
#define VOL    (160 * 160 * 160)    // 4096000

typedef __attribute__((ext_vector_type(16))) _Float16 v16h;
typedef __attribute__((ext_vector_type(8)))  float    v8f;

// Tensor Data Mover builtin exists on both toolchains with different arity:
// ROCm 7.2 / clang-22 -> 5 args, amdgpu-toolchain / clang-23 -> 6 args
// (extra zero-filled int32x8 group, probe-verified).
#if __has_builtin(__builtin_amdgcn_tensor_load_to_lds)
#define USE_TDM 1
typedef __attribute__((ext_vector_type(4))) unsigned int v4u_t;
typedef __attribute__((ext_vector_type(8))) int          v8i_t;
typedef __attribute__((ext_vector_type(4))) int          v4i_t;
#else
#define USE_TDM 0
#endif

// ---------------------------------------------------------------------------
// 1D Gaussian convolution along one axis as ONE WMMA per 16x16 tile.
//   out[o][n] = sum_k A[o][k] * X[k][n],  A[o][k] = w[k-o] (banded, K=32 pad)
//   X[k][n]  = src[u0 + k - R along conv axis, n0 + n, ow]   (zero padded)
// Strides are template constants -> immediate-offset VMEM, vectorized loads.
// A layout: 16x32 f16, same K order as B; B: 32x16 f16 (lanes 0-15 K=0..15,
// lanes 16-31 K=16..31, element e -> K = 16*hi + e).
// ---------------------------------------------------------------------------
template <int R, int SU, int SN, int SO>
__global__ __launch_bounds__(256) void blur_axis_wmma(
    const float* __restrict__ src, float* __restrict__ dst, float sigma) {
  constexpr int L = 2 * R + 1;                 // 7 (sigma=1) or 13 (sigma=2)
  const int lane = threadIdx.x & 31;
  const int wv   = threadIdx.x >> 5;
  int tile = blockIdx.x * 8 + wv;              // 2000 blocks * 8 waves = 16000
  const int ow = tile % NVOX;  tile /= NVOX;   // "other" axis coordinate
  const int tn = tile % (NVOX / 16);
  const int tu = tile / (NVOX / 16);
  const int u0 = tu * 16, n0 = tn * 16;

  const float inv2s2 = 1.0f / (2.0f * sigma * sigma);
  float wsum = 0.0f;
#pragma unroll
  for (int j = 0; j < L; ++j) {
    float d = (float)(j - R);
    wsum += __expf(-d * d * inv2s2);
  }
  const float winv = 1.0f / wsum;

  const int hi = lane >> 4;                    // half-wave select
  const int m  = lane & 15;                    // A row == output index o
  const int n  = lane & 15;                    // B column

  // ---- A fragment: banded normalized Gaussian -----------------------------
  v16h a;
#pragma unroll
  for (int e = 0; e < 16; ++e) {
    const int k = 16 * hi + e;
    const int j = k - m;                       // tap index
    float w = 0.0f;
    if (j >= 0 && j < L) {
      const float d = (float)(j - R);
      w = __expf(-d * d * inv2s2) * winv;
    }
    a[e] = (_Float16)w;
  }

  // ---- B fragment: input tile with halo, zero 'same' padding --------------
  const int base   = (n0 + n) * SN + ow * SO;
  const int ustart = u0 - R + 16 * hi;
  v16h b;
  if (u0 - R >= 0 && u0 + 31 - R < NVOX) {
    // interior tile (wave-uniform branch): unconditional, vectorizable loads
    const float* __restrict__ p = src + base + ustart * SU;
    __builtin_prefetch(p, 0, 1);               // global_prefetch_b8
#pragma unroll
    for (int e = 0; e < 16; ++e) b[e] = (_Float16)p[e * SU];
  } else {
    // edge tile: clamp address (always legal) + branch-free select-to-zero
#pragma unroll
    for (int e = 0; e < 16; ++e) {
      const int u  = ustart + e;
      const int uc = u < 0 ? 0 : (u > NVOX - 1 ? NVOX - 1 : u);
      float x = src[base + uc * SU];
      x = (u == uc) ? x : 0.0f;
      b[e] = (_Float16)x;
    }
  }

  v8f c = {};
  c = __builtin_amdgcn_wmma_f32_16x16x32_f16(
      /*neg_a=*/false, a, /*neg_b=*/false, b,
      /*c_mod=*/(short)0, c, /*reuse_a=*/false, /*reuse_b=*/false);

  // ---- D store: VGPR g -> row o = g + 8*hi, col n -------------------------
#pragma unroll
  for (int g = 0; g < 8; ++g) {
    const int o = g + 8 * hi;
    dst[(u0 + o) * SU + base] = c[g];
  }
}

// ---------------------------------------------------------------------------
// Closed-form eigenvalues of symmetric 3x3, sorted ascending by |lambda|.
// ---------------------------------------------------------------------------
__device__ __forceinline__ void eig3_abs_sorted(
    float fxx, float fyy, float fzz, float fxy, float fxz, float fyz,
    float& l1, float& l2, float& l3) {
  const float q  = (fxx + fyy + fzz) * (1.0f / 3.0f);
  const float p1 = fxy * fxy + fxz * fxz + fyz * fyz;
  const float aa = fxx - q, bb = fyy - q, cc = fzz - q;
  const float p2 = aa * aa + bb * bb + cc * cc + 2.0f * p1;
  float e0, e1, e2;
  if (p2 <= 1e-20f) {
    e0 = e1 = e2 = q;
  } else {
    const float p  = sqrtf(p2 * (1.0f / 6.0f));
    const float ip = 1.0f / p;
    const float b00 = aa * ip, b11 = bb * ip, b22 = cc * ip;
    const float b01 = fxy * ip, b02 = fxz * ip, b12 = fyz * ip;
    float detB = b00 * (b11 * b22 - b12 * b12)
               - b01 * (b01 * b22 - b12 * b02)
               + b02 * (b01 * b12 - b11 * b02);
    float r = fminf(1.0f, fmaxf(-1.0f, 0.5f * detB));
    const float phi = acosf(r) * (1.0f / 3.0f);
    e0 = q + 2.0f * p * __cosf(phi);                       // largest
    e2 = q + 2.0f * p * __cosf(phi + 2.0943951023931953f); // smallest
    e1 = 3.0f * q - e0 - e2;
  }
  float t;
  if (fabsf(e0) > fabsf(e1)) { t = e0; e0 = e1; e1 = t; }
  if (fabsf(e1) > fabsf(e2)) { t = e1; e1 = e2; e2 = t; }
  if (fabsf(e0) > fabsf(e1)) { t = e0; e0 = e1; e1 = t; }
  l1 = e0; l2 = e1; l3 = e2;
}

// ---------------------------------------------------------------------------
// Hessian (3-tap stencils on blurred volume) -> eig -> surfaceness,
// plus per-volume max via block reduce + int-atomicMax (values >= 0).
// Interior voxels take an unconditional-load fast path (no branch ladders).
// ---------------------------------------------------------------------------
__global__ __launch_bounds__(256) void hessian_surf(
    const float* __restrict__ sv, float* __restrict__ surf,
    float s2, float* __restrict__ gmax) {
  const int idx = blockIdx.x * 256 + threadIdx.x;  // grid = 16000 blocks
  const int w = idx % NVOX;
  const int t = idx / NVOX;
  const int h = t % NVOX;
  const int d = t / NVOX;

  float fxx, fyy, fzz, fxy, fxz, fyz;
  if (d >= 1 && d <= NVOX - 2 && h >= 1 && h <= NVOX - 2 &&
      w >= 1 && w <= NVOX - 2) {
    const float* __restrict__ p = sv + idx;
    const float c0 = p[0];
    fxx = (p[-SLICE] - 2.0f * c0 + p[SLICE]) * s2;
    fyy = (p[-NVOX]  - 2.0f * c0 + p[NVOX])  * s2;
    fzz = (p[-1]     - 2.0f * c0 + p[1])     * s2;
    fxy = 0.25f * (p[SLICE + NVOX] - p[-SLICE + NVOX]
                 - p[SLICE - NVOX] + p[-SLICE - NVOX]) * s2;
    fxz = 0.25f * (p[SLICE + 1] - p[-SLICE + 1]
                 - p[SLICE - 1] + p[-SLICE - 1]) * s2;
    fyz = 0.25f * (p[NVOX + 1] - p[-NVOX + 1]
                 - p[NVOX - 1] + p[-NVOX - 1]) * s2;
  } else {
    auto at = [&](int dd, int hh, int ww) -> float {
      if ((unsigned)dd >= NVOX || (unsigned)hh >= NVOX || (unsigned)ww >= NVOX)
        return 0.0f;                              // zero 'same' padding
      return sv[dd * SLICE + hh * NVOX + ww];
    };
    const float c0 = at(d, h, w);
    fxx = (at(d - 1, h, w) - 2.0f * c0 + at(d + 1, h, w)) * s2;
    fyy = (at(d, h - 1, w) - 2.0f * c0 + at(d, h + 1, w)) * s2;
    fzz = (at(d, h, w - 1) - 2.0f * c0 + at(d, h, w + 1)) * s2;
    fxy = 0.25f * (at(d + 1, h + 1, w) - at(d - 1, h + 1, w)
                 - at(d + 1, h - 1, w) + at(d - 1, h - 1, w)) * s2;
    fxz = 0.25f * (at(d + 1, h, w + 1) - at(d - 1, h, w + 1)
                 - at(d + 1, h, w - 1) + at(d - 1, h, w - 1)) * s2;
    fyz = 0.25f * (at(d, h + 1, w + 1) - at(d, h - 1, w + 1)
                 - at(d, h + 1, w - 1) + at(d, h - 1, w - 1)) * s2;
  }

  float l1, l2, l3;
  eig3_abs_sorted(fxx, fyy, fzz, fxy, fxz, fyz, l1, l2, l3);
  const float sf =
      __expf(-(l1 * l1 + l2 * l2) / (0.25f * (l3 * l3 + 1e-7f))) * fabsf(l3);
  surf[idx] = sf;

  __shared__ float red[256];
  red[threadIdx.x] = sf;
  __syncthreads();
#pragma unroll
  for (int s = 128; s > 0; s >>= 1) {
    if (threadIdx.x < s)
      red[threadIdx.x] = fmaxf(red[threadIdx.x], red[threadIdx.x + s]);
    __syncthreads();
  }
  if (threadIdx.x == 0) atomicMax((int*)gmax, __float_as_int(red[0]));
}

// Normalize by per-volume max and max-accumulate over sigmas.
__global__ __launch_bounds__(256) void norm_acc(
    const float* __restrict__ surf, const float* __restrict__ gmax,
    float* __restrict__ tot, int first) {
  const int i = blockIdx.x * 256 + threadIdx.x;
  const float mx = *gmax;
  const float v = (mx > 0.0f) ? surf[i] / (mx + 1e-7f) : 0.0f;
  tot[i] = first ? v : fmaxf(tot[i], v);
}

// Strong-threshold seed mask.
__global__ __launch_bounds__(256) void hyst_init(
    const float* __restrict__ outs, const float* __restrict__ tot,
    float* __restrict__ mask) {
  const int i = blockIdx.x * 256 + threadIdx.x;
  mask[i] = (outs[i] * tot[i] >= 0.9f) ? 1.0f : 0.0f;
}

// ---------------------------------------------------------------------------
// One hysteresis "super-pass": 8^3 tile + 1-voxel halo in LDS, iterate
// new = weak * dilate3x3x3(mask) to a local fixpoint, double-buffered.
// Interior tiles fetch the 10x10x10 `cur` block via the Tensor Data Mover
// (3D D#: tile 10x10x10 of a 160^3 fp32 tensor), waited with TENSORcnt.
// ---------------------------------------------------------------------------
__global__ __launch_bounds__(512) void hyst_pass(
    const float* __restrict__ outs, const float* __restrict__ tot,
    const float* __restrict__ cur, float* __restrict__ next) {
  __shared__ float wk[1000];
  __shared__ float m0[1000];
  __shared__ float m1[1000];

  int b = blockIdx.x;                 // 20*20*20 = 8000 tiles
  const int bx = b % 20;  b /= 20;
  const int by = b % 20;  b /= 20;
  const int bz = b;
  const int z0 = bz * 8, y0 = by * 8, x0 = bx * 8;
  const int tid = threadIdx.x;        // 512 threads = 16 waves

  // weak mask for the full halo tile (all waves)
  for (int i = tid; i < 1000; i += 512) {
    const int lx = i % 10;
    const int tt = i / 10;
    const int ly = tt % 10;
    const int lz = tt / 10;
    const int gz = z0 + lz - 1, gy = y0 + ly - 1, gx = x0 + lx - 1;
    float v = 0.0f;
    if ((unsigned)gz < NVOX && (unsigned)gy < NVOX && (unsigned)gx < NVOX) {
      const int g = gz * SLICE + gy * NVOX + gx;
      v = (outs[g] * tot[g] >= 0.5f) ? 1.0f : 0.0f;
    }
    wk[i] = v;
  }

  const bool interior = (bx > 0 && bx < 19) && (by > 0 && by < 19) &&
                        (bz > 0 && bz < 19);
  (void)interior;
#if USE_TDM
  if (interior) {
    if (tid < 32) {  // one wave issues the DMA (TDM ignores EXEC; 1 issue/wave)
      const unsigned lds = (unsigned)(size_t)(&m0[0]);
      const unsigned long long ga = (unsigned long long)(size_t)(
          cur + ((z0 - 1) * SLICE + (y0 - 1) * NVOX + (x0 - 1)));
      const v4u_t g0 = {
          1u,                                   // count=1, user descriptor
          lds,                                  // lds_addr (bytes)
          (unsigned)(ga & 0xffffffffu),         // global_addr[31:0]
          (unsigned)(ga >> 32) | (2u << 30)};   // global_addr[56:32] | type=2
      const v8i_t g1 = {
          (int)(2u << 16),                      // data_size = 4B
          (int)(160u << 16),                    // tensor_dim0[15:0]
          (int)(160u << 16),                    // dim0 hi=0 | tensor_dim1 lo16
          (int)(10u << 16),                     // dim1 hi=0 | tile_dim0 = 10
          (int)(10u | (10u << 16)),             // tile_dim1=10, tile_dim2=10
          160,                                  // tensor_dim0_stride lo32
          (int)(25600u << 16),                  // str0 hi=0 | str1[15:0]
          (int)(25600u >> 16)};                 // tensor_dim1_stride[47:16]
      const v4i_t g2 = {160, 0, SLICE * NVOX, 0};  // tensor_dim2, dim2_stride
      const v4i_t g3 = {0, 0, 0, 0};
#if defined(__clang_major__) && (__clang_major__ >= 23)
      const v8i_t gz8 = {0, 0, 0, 0, 0, 0, 0, 0};
      __builtin_amdgcn_tensor_load_to_lds(g0, g1, g2, g3, gz8, 0);
#else
      __builtin_amdgcn_tensor_load_to_lds(g0, g1, g2, g3, 0);
#endif
      __builtin_amdgcn_s_wait_tensorcnt(0);
    }
    __syncthreads();
    for (int i = tid; i < 1000; i += 512) m1[i] = m0[i];  // mirror halos
    __syncthreads();
  } else
#endif
  {
    for (int i = tid; i < 1000; i += 512) {
      const int lx = i % 10;
      const int tt = i / 10;
      const int ly = tt % 10;
      const int lz = tt / 10;
      const int gz = z0 + lz - 1, gy = y0 + ly - 1, gx = x0 + lx - 1;
      float mm = 0.0f;
      if ((unsigned)gz < NVOX && (unsigned)gy < NVOX && (unsigned)gx < NVOX)
        mm = cur[gz * SLICE + gy * NVOX + gx];
      m0[i] = mm;
      m1[i] = mm;                     // halos identical in both buffers
    }
    __syncthreads();
  }

  const int lx = tid & 7, ly = (tid >> 3) & 7, lz = tid >> 6;
  const int ci = (lz + 1) * 100 + (ly + 1) * 10 + (lx + 1);
  const float wkc = wk[ci];

#pragma unroll
  for (int it = 0; it < 10; ++it) {   // even count -> result lands in m0
    const float* src = (it & 1) ? m1 : m0;
    float*       dst = (it & 1) ? m0 : m1;
    float mx = 0.0f;
#pragma unroll
    for (int dz = -1; dz <= 1; ++dz)
#pragma unroll
      for (int dy = -1; dy <= 1; ++dy)
#pragma unroll
        for (int dx = -1; dx <= 1; ++dx)
          mx = fmaxf(mx, src[ci + dz * 100 + dy * 10 + dx]);
    dst[ci] = wkc * mx;
    __syncthreads();
  }

  next[(z0 + lz) * SLICE + (y0 + ly) * NVOX + (x0 + lx)] = m0[ci];
}

// z-only closing (window (3,1,1)): dilate pads 0, erode pads +inf (skip OOB).
__global__ __launch_bounds__(256) void zdilate(
    const float* __restrict__ m, float* __restrict__ o) {
  const int i = blockIdx.x * 256 + threadIdx.x;
  const int d = i / SLICE;
  float r = m[i];
  if (d > 0)        r = fmaxf(r, m[i - SLICE]);
  if (d < NVOX - 1) r = fmaxf(r, m[i + SLICE]);
  o[i] = r;
}
__global__ __launch_bounds__(256) void zerode(
    const float* __restrict__ m, float* __restrict__ o) {
  const int i = blockIdx.x * 256 + threadIdx.x;
  const int d = i / SLICE;
  float r = m[i];
  if (d > 0)        r = fminf(r, m[i - SLICE]);
  if (d < NVOX - 1) r = fminf(r, m[i + SLICE]);
  o[i] = r;
}

// ---------------------------------------------------------------------------
// ws layout: bufA (VOL f32) | bufB (VOL f32) | gmax[2]   => ~32.8 MB needed.
// d_out doubles as the surf_total buffer until the final erode overwrites it.
// ---------------------------------------------------------------------------
extern "C" void kernel_launch(void* const* d_in, const int* in_sizes, int n_in,
                              void* d_out, int out_size, void* d_ws,
                              size_t ws_size, hipStream_t stream) {
  const float* outs = (const float*)d_in[0];
  float* out  = (float*)d_out;
  float* bufA = (float*)d_ws;
  float* bufB = bufA + VOL;
  float* gmax = bufB + VOL;           // 2 scalars (one per sigma)
  float* tot  = out;                  // surf_total lives in d_out until end

  hipMemsetAsync(gmax, 0, 2 * sizeof(float), stream);

  const dim3 tb(256);
  const int TG = 2000;                // 2000 blocks * 8 waves = 16000 tiles
  const int VG = VOL / 256;           // 16000 blocks, 1 thread per voxel

  // ---- sigma = 1 (R = 3): blur D -> H -> W, then Hessian/eig/surf ----
  blur_axis_wmma<3, SLICE, 1, NVOX><<<TG, tb, 0, stream>>>(outs, bufA, 1.0f);
  blur_axis_wmma<3, NVOX, 1, SLICE><<<TG, tb, 0, stream>>>(bufA, bufB, 1.0f);
  blur_axis_wmma<3, 1, NVOX, SLICE><<<TG, tb, 0, stream>>>(bufB, bufA, 1.0f);
  hessian_surf<<<VG, tb, 0, stream>>>(bufA, bufB, 1.0f, gmax);
  norm_acc<<<VG, tb, 0, stream>>>(bufB, gmax, tot, 1);

  // ---- sigma = 2 (R = 6) ----
  blur_axis_wmma<6, SLICE, 1, NVOX><<<TG, tb, 0, stream>>>(outs, bufA, 2.0f);
  blur_axis_wmma<6, NVOX, 1, SLICE><<<TG, tb, 0, stream>>>(bufA, bufB, 2.0f);
  blur_axis_wmma<6, 1, NVOX, SLICE><<<TG, tb, 0, stream>>>(bufB, bufA, 2.0f);
  hessian_surf<<<VG, tb, 0, stream>>>(bufA, bufB, 4.0f, gmax + 1);
  norm_acc<<<VG, tb, 0, stream>>>(bufB, gmax + 1, tot, 0);

  // ---- hysteresis: seed with strong, then LDS-tile fixpoint passes ----
  hyst_init<<<VG, tb, 0, stream>>>(outs, tot, bufA);
  for (int p = 0; p < 10; ++p) {      // even count -> final mask in bufA
    const float* c = (p & 1) ? bufB : bufA;
    float*      nx = (p & 1) ? bufA : bufB;
    hyst_pass<<<8000, 512, 0, stream>>>(outs, tot, c, nx);
  }

  // ---- anisotropic closing along z (win (3,1,1)) ----
  zdilate<<<VG, tb, 0, stream>>>(bufA, bufB);
  zerode<<<VG, tb, 0, stream>>>(bufB, out);
}
